// Quantized4BLinearLayer_18202071400828
// MI455X (gfx1250) — matrix operand order repaired and checked by reading, not compiled, not run on hardware
//
#include <hip/hip_runtime.h>

typedef __attribute__((ext_vector_type(16))) _Float16 v16h;
typedef __attribute__((ext_vector_type(8)))  _Float16 v8h;
typedef __attribute__((ext_vector_type(4)))  _Float16 v4h;
typedef __attribute__((ext_vector_type(8)))  float    v8f;
typedef __attribute__((ext_vector_type(4)))  float    v4f;
typedef __attribute__((ext_vector_type(4)))  int      v4i;
typedef __attribute__((ext_vector_type(8)))  unsigned v8u;

constexpr int TOKENS = 32;
constexpr int INF    = 8192;
constexpr int OUTF   = 8192;
constexpr int KPACK  = INF / 2;          // packed int32 "bytes" per output row
constexpr int WAVES  = 4;                // waves per block
constexpr int BLOCK  = WAVES * 32;       // wave32
constexpr int COLS_PER_BLOCK = WAVES * 16;

// ---- one-shot: x (f32, row-major [32,8192]) -> f16 in workspace ----
__global__ void __launch_bounds__(256)
x_to_f16(const float* __restrict__ x, _Float16* __restrict__ xh) {
    int i = (blockIdx.x * blockDim.x + threadIdx.x) * 4;
    v4f v = *(const v4f*)(x + i);
    v4h h;
    h[0] = (_Float16)v[0];
    h[1] = (_Float16)v[1];
    h[2] = (_Float16)v[2];
    h[3] = (_Float16)v[3];
    *(v4h*)(xh + i) = h;
}

// Decode one packed byte (int32 holding a value in [0,256)) into a dword of
// two f16 codes: [15:0] = f16(1024 + hi_nibble), [31:16] = f16(1024 + lo_nibble).
// Subtracting 1032.0h afterwards yields exactly (nibble - 8).
__device__ __forceinline__ unsigned nib2h(unsigned b) {
    return (((b & 15u) << 16) | ((b >> 4) & 15u)) | 0x64006400u;
}

// ---- main kernel: int4 dequant GEMM via v_wmma_f32_16x16x32_f16 ----
// Each wave: 16 output columns, full K sweep, both 16-token M tiles.
template<bool XH>
__global__ void __launch_bounds__(BLOCK)
q4_wmma(const float* __restrict__ xf,
        const _Float16* __restrict__ xh,
        const int* __restrict__ pw,
        const float* __restrict__ scale,
        const float* __restrict__ bias,
        float* __restrict__ out)
{
    const int tid  = threadIdx.x;
    const int lane = tid & 31;
    const int wave = tid >> 5;
    const int n    = lane & 15;   // column-within-tile (B/N, C/D col, A row M%16)
    const int g    = lane >> 4;   // lane half selector
    const int o    = blockIdx.x * COLS_PER_BLOCK + wave * 16 + n;  // output col

    const int* __restrict__ wrow = pw + (size_t)o * KPACK;

    v8f c0 = {};   // tokens 0..15
    v8f c1 = {};   // tokens 16..31

    const size_t m0 = (size_t)n;
    const size_t m1 = (size_t)(n + 16);

    // packed f16 bias: -1032.0 in every element (v_pk_add_f16 x8)
    v16h hbias;
#pragma unroll
    for (int e = 0; e < 16; ++e) hbias[e] = (_Float16)(-1032.0f);

#pragma unroll 2
    for (int k0 = 0; k0 < INF; k0 += 32) {
        // ---------- B fragment: column o, K = kb .. kb+15 (16 consecutive) ----------
        const int kb = k0 + g * 16;
        const int* bp = wrow + (kb >> 1);
        v4i p0 = *(const v4i*)bp;         // K pairs (kb+0,1)..(kb+6,7)
        v4i p1 = *(const v4i*)(bp + 4);   // K pairs (kb+8,9)..(kb+14,15)
        __builtin_prefetch(bp + 16 * 8, 0, 1);   // weight stream, 8 iters ahead

        v8u braw;
#pragma unroll
        for (int p = 0; p < 4; ++p) {
            braw[p]     = nib2h((unsigned)p0[p]);   // elements 2p, 2p+1
            braw[p + 4] = nib2h((unsigned)p1[p]);   // elements 8+2p, 9+2p
        }
        v16h bfrag = __builtin_bit_cast(v16h, braw);
        bfrag = bfrag + hbias;                      // (1024+n) - 1032 = n - 8, exact

        // ---------- A fragments: token m, K runs [ka0,ka0+8) and [ka0+8,+8) ----------
        const int ka0 = k0 + g * 16;
        const int ka1 = ka0 + 8;
        v16h a0, a1;
        if constexpr (XH) {
            v8h t0 = *(const v8h*)(xh + m0 * INF + ka0);
            v8h t1 = *(const v8h*)(xh + m0 * INF + ka1);
            v8h t2 = *(const v8h*)(xh + m1 * INF + ka0);
            v8h t3 = *(const v8h*)(xh + m1 * INF + ka1);
#pragma unroll
            for (int e = 0; e < 8; ++e) {
                a0[e] = t0[e];  a0[8 + e] = t1[e];
                a1[e] = t2[e];  a1[8 + e] = t3[e];
            }
        } else {
            v4f f0 = *(const v4f*)(xf + m0 * INF + ka0);
            v4f f1 = *(const v4f*)(xf + m0 * INF + ka0 + 4);
            v4f f2 = *(const v4f*)(xf + m0 * INF + ka1);
            v4f f3 = *(const v4f*)(xf + m0 * INF + ka1 + 4);
            v4f f4 = *(const v4f*)(xf + m1 * INF + ka0);
            v4f f5 = *(const v4f*)(xf + m1 * INF + ka0 + 4);
            v4f f6 = *(const v4f*)(xf + m1 * INF + ka1);
            v4f f7 = *(const v4f*)(xf + m1 * INF + ka1 + 4);
#pragma unroll
            for (int e = 0; e < 4; ++e) {
                a0[e]      = (_Float16)f0[e];
                a0[4 + e]  = (_Float16)f1[e];
                a0[8 + e]  = (_Float16)f2[e];
                a0[12 + e] = (_Float16)f3[e];
                a1[e]      = (_Float16)f4[e];
                a1[4 + e]  = (_Float16)f5[e];
                a1[8 + e]  = (_Float16)f6[e];
                a1[12 + e] = (_Float16)f7[e];
            }
        }

        // D = A x B + C   (16x16x32 f16 -> f32 accum)
        c0 = __builtin_amdgcn_wmma_f32_16x16x32_f16(
                false, a0, false, bfrag, (short)0, c0, false, false);
        c1 = __builtin_amdgcn_wmma_f32_16x16x32_f16(
                false, a1, false, bfrag, (short)0, c1, false, false);
    }

    // ---------- epilogue: y = scale[o] * acc + bias[o] ----------
    const float s  = scale[o];   // scale is [OUT,1] flattened
    const float bi = bias[o];
#pragma unroll
    for (int r = 0; r < 8; ++r) {
        const int mrow = r + 8 * g;  // C/D layout: VGPR r -> row r + 8*(lane/16)
        out[(size_t)mrow * OUTF + o]        = c0[r] * s + bi;
        out[(size_t)(mrow + 16) * OUTF + o] = c1[r] * s + bi;
    }
}

extern "C" void kernel_launch(void* const* d_in, const int* in_sizes, int n_in,
                              void* d_out, int out_size, void* d_ws, size_t ws_size,
                              hipStream_t stream) {
    const float* x  = (const float*)d_in[0];
    const int*   pw = (const int*)d_in[1];
    const float* sc = (const float*)d_in[2];
    const float* bs = (const float*)d_in[3];
    float* out = (float*)d_out;

    const size_t xh_bytes = (size_t)TOKENS * INF * 2;
    const dim3 grid(OUTF / COLS_PER_BLOCK);   // 128 blocks x 4 waves

    if (ws_size >= xh_bytes) {
        _Float16* xh = (_Float16*)d_ws;
        const int nconv = TOKENS * INF / 4;   // 4 elements per thread
        x_to_f16<<<nconv / 256, 256, 0, stream>>>(x, xh);
        q4_wmma<true><<<grid, BLOCK, 0, stream>>>(x, xh, pw, sc, bs, out);
    } else {
        q4_wmma<false><<<grid, BLOCK, 0, stream>>>(x, nullptr, pw, sc, bs, out);
    }
}
